// MultiHeadedAttention_25744033973041
// MI455X (gfx1250) — matrix operand order repaired and checked by reading, not compiled, not run on hardware
//
#include <hip/hip_runtime.h>
#include <hip/hip_bf16.h>

#define BB 2
#define SS 2048
#define HH 12
#define DM 768
#define DK 64

typedef _Float16 v16h __attribute__((ext_vector_type(16)));
typedef _Float16 v8h  __attribute__((ext_vector_type(8)));
typedef _Float16 v4h  __attribute__((ext_vector_type(4)));
typedef float    v8f  __attribute__((ext_vector_type(8)));

#define WMMA_F16(a, b, c) \
    __builtin_amdgcn_wmma_f32_16x16x32_f16(false, (a), false, (b), (short)0, (c), false, false)

// ---------------------------------------------------------------------------
// fp32 -> f16 conversion (n divisible by 4)
// ---------------------------------------------------------------------------
__global__ void f32_to_f16_kernel(const float* __restrict__ src,
                                  _Float16* __restrict__ dst, int n) {
    int i = (blockIdx.x * blockDim.x + threadIdx.x) * 4;
    if (i < n) {
        float4 v = *(const float4*)(src + i);
        v4h h;
        h[0] = (_Float16)v.x; h[1] = (_Float16)v.y;
        h[2] = (_Float16)v.z; h[3] = (_Float16)v.w;
        *(v4h*)(dst + i) = h;
    }
}

// ---------------------------------------------------------------------------
// Fragment loaders (ISA 7.12.2 layouts).
// A 16x32 f16: lane holds M = lane&15; K chunks [k0+8*half .. +7] and
//              [k0+16+8*half .. +7]  (half = lane>>4)
// B 32x16 f16: lane holds N = lane&15; same K chunks as A
// ---------------------------------------------------------------------------
__device__ __forceinline__ v16h ldfragA(const _Float16* p, int ld, int rbase, int k0) {
    int lane = threadIdx.x & 31;
    const _Float16* base = p + (rbase + (lane & 15)) * ld + k0 + ((lane >> 4) << 3);
    v8h lo = *(const v8h*)(base);
    v8h hi = *(const v8h*)(base + 16);
    return __builtin_shufflevector(lo, hi, 0, 1, 2, 3, 4, 5, 6, 7,
                                           8, 9, 10, 11, 12, 13, 14, 15);
}

__device__ __forceinline__ v16h ldfragB(const _Float16* p, int ld, int rbase, int k0) {
    int lane = threadIdx.x & 31;
    const _Float16* base = p + (rbase + (lane & 15)) * ld + k0 + ((lane >> 4) << 3);
    v8h lo = *(const v8h*)(base);
    v8h hi = *(const v8h*)(base + 16);
    return __builtin_shufflevector(lo, hi, 0, 1, 2, 3, 4, 5, 6, 7,
                                           8, 9, 10, 11, 12, 13, 14, 15);
}

// ---------------------------------------------------------------------------
// Projection GEMM: Y[m][n] = sum_k X[m][k] * W[n][k] + bias[n]
// M = B*S = 4096, N = K = 768.  Block: 256 thr = 8 waves (2x4), wave tile 32x64.
// mode 0: store as [B][H][S][DK] (Q/K);  mode 1: store V transposed [B][H][DK][S]
// ---------------------------------------------------------------------------
__global__ __launch_bounds__(256) void proj_gemm_kernel(
        const _Float16* __restrict__ X, const _Float16* __restrict__ W,
        const float* __restrict__ bias, _Float16* __restrict__ out, int mode) {
    const int lane = threadIdx.x & 31;
    const int wave = threadIdx.x >> 5;
    const int m0 = blockIdx.x * 64 + (wave >> 2) * 32;
    const int n0 = blockIdx.y * 256 + (wave & 3) * 64;

    v8f acc[2][4];
#pragma unroll
    for (int mi = 0; mi < 2; ++mi)
#pragma unroll
        for (int ni = 0; ni < 4; ++ni) acc[mi][ni] = (v8f)0.0f;

    for (int k0 = 0; k0 < DM; k0 += 32) {
        v16h a[2], b[4];
#pragma unroll
        for (int mi = 0; mi < 2; ++mi) a[mi] = ldfragA(X, DM, m0 + mi * 16, k0);
#pragma unroll
        for (int ni = 0; ni < 4; ++ni) b[ni] = ldfragB(W, DM, n0 + ni * 16, k0);
#pragma unroll
        for (int mi = 0; mi < 2; ++mi)
#pragma unroll
            for (int ni = 0; ni < 4; ++ni)
                acc[mi][ni] = WMMA_F16(a[mi], b[ni], acc[mi][ni]);
    }

    const int half = lane >> 4;
#pragma unroll
    for (int mi = 0; mi < 2; ++mi) {
#pragma unroll
        for (int ni = 0; ni < 4; ++ni) {
            int col = n0 + ni * 16 + (lane & 15);
            float bv = bias[col];
            int h = col >> 6, d = col & 63;
#pragma unroll
            for (int j = 0; j < 8; ++j) {
                int row = m0 + mi * 16 + half * 8 + j;     // = b*S + s
                int bidx = row >> 11, s = row & (SS - 1);
                float y = acc[mi][ni][j] + bv;
                int idx = mode ? (((bidx * HH + h) * DK + d) * SS + s)
                               : (((bidx * HH + h) * SS + s) * DK + d);
                out[idx] = (_Float16)y;
            }
        }
    }
}

// ---------------------------------------------------------------------------
// Flash attention.  1 wave = 32 q rows; block = 4 waves = 128 q rows.
// Qh/Kh: [B][H][S][64] f16,  Vt: [B][H][64][S] f16,  ctx: [B][S][768] f16
// ---------------------------------------------------------------------------
__global__ __launch_bounds__(128, 1) void flash_attn_kernel(
        const _Float16* __restrict__ Qh, const _Float16* __restrict__ Kh,
        const _Float16* __restrict__ Vt, _Float16* __restrict__ ctx) {
    const int lane = threadIdx.x & 31;
    const int wave = threadIdx.x >> 5;
    const int half = lane >> 4;
    const int b = blockIdx.z, h = blockIdx.y;
    const int q0 = blockIdx.x * 128 + wave * 32;

    const _Float16* Qp = Qh + (b * HH + h) * (SS * DK);
    const _Float16* Kp = Kh + (b * HH + h) * (SS * DK);
    const _Float16* Vp = Vt + (b * HH + h) * (DK * SS);

    __shared__ _Float16 sP[4][32][48];   // per-wave 32x32 P tile, padded stride

    // Q fragments stay resident: [m-subtile][d-chunk of 32]
    v16h qf[2][2];
#pragma unroll
    for (int mi = 0; mi < 2; ++mi)
#pragma unroll
        for (int kk = 0; kk < 2; ++kk)
            qf[mi][kk] = ldfragA(Qp, DK, q0 + mi * 16, kk * 32);

    float mrow[2][8], lrow[2][8];
    v8f oacc[2][4];
#pragma unroll
    for (int mi = 0; mi < 2; ++mi) {
#pragma unroll
        for (int j = 0; j < 8; ++j) { mrow[mi][j] = -1e30f; lrow[mi][j] = 0.0f; }
#pragma unroll
        for (int ni = 0; ni < 4; ++ni) oacc[mi][ni] = (v8f)0.0f;
    }

    const float scale = 0.125f;  // 1/sqrt(64)

    for (int kb = 0; kb < SS; kb += 32) {
        // ---- scores S = Q * K^T  (B operand: lane n -> row K[kb+n][:]) ----
        v16h kf[2][2];
#pragma unroll
        for (int ni = 0; ni < 2; ++ni)
#pragma unroll
            for (int kk = 0; kk < 2; ++kk)
                kf[ni][kk] = ldfragB(Kp, DK, kb + ni * 16, kk * 32);

        v8f sc[2][2];
#pragma unroll
        for (int mi = 0; mi < 2; ++mi)
#pragma unroll
            for (int ni = 0; ni < 2; ++ni) {
                v8f c = (v8f)0.0f;
                c = WMMA_F16(qf[mi][0], kf[ni][0], c);
                c = WMMA_F16(qf[mi][1], kf[ni][1], c);
                sc[mi][ni] = c;
            }

        // ---- online softmax (rows: element j <-> row j + 8*half) ----
#pragma unroll
        for (int mi = 0; mi < 2; ++mi) {
            float tmax[8];
#pragma unroll
            for (int j = 0; j < 8; ++j)
                tmax[j] = fmaxf(sc[mi][0][j], sc[mi][1][j]) * scale;
#pragma unroll
            for (int off = 1; off < 16; off <<= 1)
#pragma unroll
                for (int j = 0; j < 8; ++j)
                    tmax[j] = fmaxf(tmax[j], __shfl_xor(tmax[j], off, 32));

            float corr[8], rsum[8], p[2][8];
#pragma unroll
            for (int j = 0; j < 8; ++j) {
                float nm = fmaxf(mrow[mi][j], tmax[j]);
                corr[j] = __expf(mrow[mi][j] - nm);
                mrow[mi][j] = nm;
                rsum[j] = 0.0f;
            }
#pragma unroll
            for (int ni = 0; ni < 2; ++ni)
#pragma unroll
                for (int j = 0; j < 8; ++j) {
                    p[ni][j] = __expf(sc[mi][ni][j] * scale - mrow[mi][j]);
                    rsum[j] += p[ni][j];
                }
#pragma unroll
            for (int off = 1; off < 16; off <<= 1)
#pragma unroll
                for (int j = 0; j < 8; ++j)
                    rsum[j] += __shfl_xor(rsum[j], off, 32);
#pragma unroll
            for (int j = 0; j < 8; ++j)
                lrow[mi][j] = lrow[mi][j] * corr[j] + rsum[j];
#pragma unroll
            for (int ni = 0; ni < 4; ++ni)
#pragma unroll
                for (int j = 0; j < 8; ++j)
                    oacc[mi][ni][j] *= corr[j];
            // P (C layout) -> LDS
#pragma unroll
            for (int ni = 0; ni < 2; ++ni)
#pragma unroll
                for (int j = 0; j < 8; ++j)
                    sP[wave][mi * 16 + half * 8 + j][ni * 16 + (lane & 15)] =
                        (_Float16)p[ni][j];
        }
        __syncthreads();

        // ---- O += P * V  (P re-read in A layout, V from transposed layout) ----
        v16h pf[2];
#pragma unroll
        for (int mi = 0; mi < 2; ++mi) {
            const _Float16* base = &sP[wave][mi * 16 + (lane & 15)][half * 8];
            v8h lo = *(const v8h*)(base);
            v8h hi = *(const v8h*)(base + 16);
            pf[mi] = __builtin_shufflevector(lo, hi, 0, 1, 2, 3, 4, 5, 6, 7,
                                                     8, 9, 10, 11, 12, 13, 14, 15);
        }
        v16h vf[4];
#pragma unroll
        for (int ni = 0; ni < 4; ++ni)
            vf[ni] = ldfragB(Vp, SS, ni * 16, kb);   // lane n -> Vt[d=n][kb..]
#pragma unroll
        for (int mi = 0; mi < 2; ++mi)
#pragma unroll
            for (int ni = 0; ni < 4; ++ni)
                oacc[mi][ni] = WMMA_F16(pf[mi], vf[ni], oacc[mi][ni]);
        __syncthreads();
    }

    // ---- normalize and store ctx f16 ----
#pragma unroll
    for (int mi = 0; mi < 2; ++mi)
#pragma unroll
        for (int ni = 0; ni < 4; ++ni) {
            int col = h * 64 + ni * 16 + (lane & 15);
#pragma unroll
            for (int j = 0; j < 8; ++j) {
                int row = q0 + mi * 16 + half * 8 + j;
                float o = oacc[mi][ni][j] / lrow[mi][j];
                ctx[(b * SS + row) * DM + col] = (_Float16)o;
            }
        }
}

// ---------------------------------------------------------------------------
// Output projection: out[m][n] = sum_k ctx[m][k] * Wo[n][k] + bo[n]  (fp32 out)
// ---------------------------------------------------------------------------
__global__ __launch_bounds__(256) void out_gemm_kernel(
        const _Float16* __restrict__ A, const _Float16* __restrict__ W,
        const float* __restrict__ bias, float* __restrict__ out) {
    const int lane = threadIdx.x & 31;
    const int wave = threadIdx.x >> 5;
    const int m0 = blockIdx.x * 64 + (wave >> 2) * 32;
    const int n0 = blockIdx.y * 256 + (wave & 3) * 64;

    v8f acc[2][4];
#pragma unroll
    for (int mi = 0; mi < 2; ++mi)
#pragma unroll
        for (int ni = 0; ni < 4; ++ni) acc[mi][ni] = (v8f)0.0f;

    for (int k0 = 0; k0 < DM; k0 += 32) {
        v16h a[2], b[4];
#pragma unroll
        for (int mi = 0; mi < 2; ++mi) a[mi] = ldfragA(A, DM, m0 + mi * 16, k0);
#pragma unroll
        for (int ni = 0; ni < 4; ++ni) b[ni] = ldfragB(W, DM, n0 + ni * 16, k0);
#pragma unroll
        for (int mi = 0; mi < 2; ++mi)
#pragma unroll
            for (int ni = 0; ni < 4; ++ni)
                acc[mi][ni] = WMMA_F16(a[mi], b[ni], acc[mi][ni]);
    }

    const int half = lane >> 4;
#pragma unroll
    for (int mi = 0; mi < 2; ++mi)
#pragma unroll
        for (int ni = 0; ni < 4; ++ni) {
            int col = n0 + ni * 16 + (lane & 15);
            float bv = bias[col];
#pragma unroll
            for (int j = 0; j < 8; ++j) {
                int row = m0 + mi * 16 + half * 8 + j;
                out[row * DM + col] = acc[mi][ni][j] + bv;
            }
        }
}

// ---------------------------------------------------------------------------
extern "C" void kernel_launch(void* const* d_in, const int* in_sizes, int n_in,
                              void* d_out, int out_size, void* d_ws, size_t ws_size,
                              hipStream_t stream) {
    (void)in_sizes; (void)n_in; (void)out_size; (void)ws_size;
    const float* query = (const float*)d_in[0];
    const float* key   = (const float*)d_in[1];
    const float* value = (const float*)d_in[2];
    const float* Wq = (const float*)d_in[3];
    const float* bq = (const float*)d_in[4];
    const float* Wk = (const float*)d_in[5];
    const float* bk = (const float*)d_in[6];
    const float* Wv = (const float*)d_in[7];
    const float* bv = (const float*)d_in[8];
    const float* Wo = (const float*)d_in[9];
    const float* bo = (const float*)d_in[10];
    float* out = (float*)d_out;

    const int XSZ = BB * SS * DM;   // 3,145,728
    const int WSZ = DM * DM;        //   589,824

    _Float16* ws  = (_Float16*)d_ws;
    _Float16* Xq  = ws;
    _Float16* Xk  = ws + XSZ;
    _Float16* Xv  = ws + 2 * XSZ;
    _Float16* Wqh = ws + 3 * XSZ;
    _Float16* Wkh = Wqh + WSZ;
    _Float16* Wvh = Wqh + 2 * WSZ;
    _Float16* Woh = Wqh + 3 * WSZ;
    _Float16* Qh  = Wqh + 4 * WSZ;
    _Float16* Kh  = Qh + XSZ;
    _Float16* Vt  = Qh + 2 * XSZ;
    _Float16* Ctx = Qh + 3 * XSZ;

    // 1) fp32 -> f16 conversions
    f32_to_f16_kernel<<<XSZ / 4 / 256, 256, 0, stream>>>(query, Xq, XSZ);
    f32_to_f16_kernel<<<XSZ / 4 / 256, 256, 0, stream>>>(key,   Xk, XSZ);
    f32_to_f16_kernel<<<XSZ / 4 / 256, 256, 0, stream>>>(value, Xv, XSZ);
    f32_to_f16_kernel<<<WSZ / 4 / 256, 256, 0, stream>>>(Wq, Wqh, WSZ);
    f32_to_f16_kernel<<<WSZ / 4 / 256, 256, 0, stream>>>(Wk, Wkh, WSZ);
    f32_to_f16_kernel<<<WSZ / 4 / 256, 256, 0, stream>>>(Wv, Wvh, WSZ);
    f32_to_f16_kernel<<<WSZ / 4 / 256, 256, 0, stream>>>(Wo, Woh, WSZ);

    // 2) Q/K/V projections (V stored transposed per head)
    dim3 ggrid(BB * SS / 64, DM / 256);
    proj_gemm_kernel<<<ggrid, 256, 0, stream>>>(Xq, Wqh, bq, Qh, 0);
    proj_gemm_kernel<<<ggrid, 256, 0, stream>>>(Xk, Wkh, bk, Kh, 0);
    proj_gemm_kernel<<<ggrid, 256, 0, stream>>>(Xv, Wvh, bv, Vt, 1);

    // 3) flash attention
    flash_attn_kernel<<<dim3(SS / 128, HH, BB), 128, 0, stream>>>(Qh, Kh, Vt, Ctx);

    // 4) output projection -> fp32
    out_gemm_kernel<<<ggrid, 256, 0, stream>>>(Ctx, Woh, bo, out);
}
